// TransformerBlock_46866683134124
// MI455X (gfx1250) — matrix operand order repaired and checked by reading, not compiled, not run on hardware
//
#include <hip/hip_runtime.h>

// ---------------------------------------------------------------------------
// Types for CDNA5 WMMA (gfx1250, wave32)
// ---------------------------------------------------------------------------
typedef __bf16 bf16_t;
typedef __attribute__((ext_vector_type(16))) __bf16 v16bf;
typedef __attribute__((ext_vector_type(8)))  float  v8f;
typedef __attribute__((ext_vector_type(4)))  float  vfloat4;

union Frag { v16bf v; vfloat4 f4[2]; };

__device__ __forceinline__ float bf2f(unsigned short h) {
    return __uint_as_float(((unsigned)h) << 16);
}
__device__ __forceinline__ unsigned short f2b(float f) {
    unsigned u = __float_as_uint(f);
    u = u + 0x7FFFu + ((u >> 16) & 1u);          // round-to-nearest-even
    return (unsigned short)(u >> 16);
}

#define HW_  16384
#define CDIM 256

// ---------------------------------------------------------------------------
// Small helpers
// ---------------------------------------------------------------------------
__global__ void convert_weight_kernel(const float* __restrict__ w,
                                      unsigned short* __restrict__ wb,
                                      int O, int K, int Opad, int Kpad) {
    long i = (long)blockIdx.x * 256 + threadIdx.x;
    long tot = (long)Opad * Kpad;
    if (i >= tot) return;
    int o = (int)(i / Kpad), k = (int)(i % Kpad);
    float v = (o < O && k < K) ? w[(long)o * K + k] : 0.f;
    wb[i] = f2b(v);
}

__global__ void pad_bias_kernel(const float* __restrict__ b,
                                float* __restrict__ bp, int O, int Opad) {
    int i = blockIdx.x * 256 + threadIdx.x;
    if (i < Opad) bp[i] = (i < O) ? b[i] : 0.f;
}

__global__ void zero_kernel(float* __restrict__ p, int n) {
    int i = blockIdx.x * 256 + threadIdx.x;
    if (i < n) p[i] = 0.f;
}

// ---------------------------------------------------------------------------
// LayerNorm over channels: NCHW f32 in -> NHWC bf16 out (R x 256)
// One thread per pixel; strided channel reads are coalesced across threads.
// ---------------------------------------------------------------------------
__global__ __launch_bounds__(256) void ln_kernel(const float* __restrict__ x,
                                                 const float* __restrict__ gamma,
                                                 const float* __restrict__ beta,
                                                 unsigned short* __restrict__ out) {
    int b = blockIdx.y;
    int n = blockIdx.x * 256 + threadIdx.x;
    const float* xp = x + (long)b * CDIM * HW_ + n;
    float s = 0.f, s2 = 0.f;
    for (int c = 0; c < CDIM; ++c) {
        float v = xp[(long)c * HW_];
        s += v; s2 += v * v;
    }
    float mu  = s * (1.f / CDIM);
    float var = s2 * (1.f / CDIM) - mu * mu;
    float inv = rsqrtf(var + 1e-5f);

    __shared__ unsigned short tile[256 * 16];
    long outBase = ((long)b * HW_ + (long)blockIdx.x * 256) * CDIM;
    for (int c0 = 0; c0 < CDIM; c0 += 16) {
#pragma unroll
        for (int cc = 0; cc < 16; ++cc) {
            float v = (xp[(long)(c0 + cc) * HW_] - mu) * inv * gamma[c0 + cc] + beta[c0 + cc];
            tile[threadIdx.x * 16 + cc] = f2b(v);
        }
        __syncthreads();
#pragma unroll
        for (int i2 = 0; i2 < 16; ++i2) {
            int idx = i2 * 256 + threadIdx.x;
            int p = idx >> 4, cc = idx & 15;
            out[outBase + (long)p * CDIM + c0 + cc] = tile[p * 16 + cc];
        }
        __syncthreads();
    }
}

// ---------------------------------------------------------------------------
// WMMA GEMM:  Y[R x Opad] = A[R x K] (NHWC bf16) * W[Opad x K]^T (bf16) + bias
// Block: 256 threads = 8 waves, tile 64 rows x 128 out-channels.
// Wave: 32x32 via 2x2 v_wmma_f32_16x16x32_bf16 accumulators.
// MODE 0: store bf16 NHWC (stride Opad)
// MODE 1: out = bias + acc + res (res/out NCHW f32, Opad == Oreal)
// ---------------------------------------------------------------------------
template <int MODE>
__global__ __launch_bounds__(256) void gemm_bf16(const unsigned short* __restrict__ A,
                                                 const unsigned short* __restrict__ Wt,
                                                 const float* __restrict__ bias,
                                                 unsigned short* __restrict__ outBf,
                                                 float* __restrict__ outF,
                                                 const float* __restrict__ res,
                                                 int K, int Opad, int Oreal) {
    int lane = threadIdx.x & 31;
    int wave = threadIdx.x >> 5;
    int half = lane >> 4;
    int l16  = lane & 15;
    int wm = wave >> 2;                 // 0..1 (row dim)
    int wo = wave & 3;                  // 0..3 (out-channel dim)

    long rowBase = (long)blockIdx.x * 64 + (long)wm * 32;
    int  oBase   = blockIdx.y * 128 + wo * 32;

    v8f acc[2][2];
#pragma unroll
    for (int mi = 0; mi < 2; ++mi)
#pragma unroll
        for (int nj = 0; nj < 2; ++nj)
            acc[mi][nj] = (v8f){0.f, 0.f, 0.f, 0.f, 0.f, 0.f, 0.f, 0.f};

    const unsigned short* aRow0 = A  + (rowBase + l16) * (long)K;
    const unsigned short* aRow1 = aRow0 + 16L * K;
    const unsigned short* bCol0 = Wt + ((long)(oBase + l16)) * K;
    const unsigned short* bCol1 = bCol0 + 16L * K;

    for (int k0 = 0; k0 < K; k0 += 32) {
        // A fragment layout (16-bit A 16x32): lane half h -> k runs [8h,8h+8) and [16+8h,16+8h+8)
        int ka = k0 + half * 8;
        Frag a0, a1, b0, b1;
        a0.f4[0] = *(const vfloat4*)(aRow0 + ka);
        a0.f4[1] = *(const vfloat4*)(aRow0 + ka + 16);
        a1.f4[0] = *(const vfloat4*)(aRow1 + ka);
        a1.f4[1] = *(const vfloat4*)(aRow1 + ka + 16);
        // B fragment layout (16-bit B 32x16): lane half h -> k runs [8h,8h+8) and [16+8h,16+8h+8)
        int kb = k0 + half * 8;
        b0.f4[0] = *(const vfloat4*)(bCol0 + kb);
        b0.f4[1] = *(const vfloat4*)(bCol0 + kb + 16);
        b1.f4[0] = *(const vfloat4*)(bCol1 + kb);
        b1.f4[1] = *(const vfloat4*)(bCol1 + kb + 16);

        __builtin_prefetch(aRow0 + k0 + 128, 0, 0);   // global_prefetch_b8
        __builtin_prefetch(aRow1 + k0 + 128, 0, 0);

        acc[0][0] = __builtin_amdgcn_wmma_f32_16x16x32_bf16(false, a0.v, false, b0.v,
                                                            (short)0, acc[0][0], false, false);
        acc[0][1] = __builtin_amdgcn_wmma_f32_16x16x32_bf16(false, a0.v, false, b1.v,
                                                            (short)0, acc[0][1], false, false);
        acc[1][0] = __builtin_amdgcn_wmma_f32_16x16x32_bf16(false, a1.v, false, b0.v,
                                                            (short)0, acc[1][0], false, false);
        acc[1][1] = __builtin_amdgcn_wmma_f32_16x16x32_bf16(false, a1.v, false, b1.v,
                                                            (short)0, acc[1][1], false, false);
    }

    // C/D layout: lane col = l16, VGPR r -> row r + 8*half
#pragma unroll
    for (int mi = 0; mi < 2; ++mi) {
#pragma unroll
        for (int nj = 0; nj < 2; ++nj) {
            int o = oBase + nj * 16 + l16;
            float bv = bias[o];
#pragma unroll
            for (int r = 0; r < 8; ++r) {
                long row = rowBase + mi * 16 + 8 * half + r;   // global pixel index
                float v = acc[mi][nj][r] + bv;
                if (MODE == 0) {
                    outBf[row * (long)Opad + o] = f2b(v);
                } else {
                    long bb = row >> 14;          // HW == 16384
                    long nn = row & 16383;
                    long idx = ((long)bb * Oreal + o) * (long)HW_ + nn;  // NCHW
                    outF[idx] = v + res[idx];
                }
            }
        }
    }
}

// ---------------------------------------------------------------------------
// Depthwise 3x3 SAME, NHWC bf16 -> bf16.  Thread = (pixel, channel).
// ---------------------------------------------------------------------------
__global__ __launch_bounds__(256) void dwconv_kernel(const unsigned short* __restrict__ in,
                                                     const float* __restrict__ w,
                                                     const float* __restrict__ bias,
                                                     unsigned short* __restrict__ out,
                                                     int Cpad, int Creal) {
    int c = blockIdx.y * 256 + threadIdx.x;
    if (c >= Cpad) return;
    int n = blockIdx.x;               // 0..HW-1
    int b = blockIdx.z;
    int h = n >> 7, x = n & 127;      // W == 128
    long base = (long)b * HW_ * Cpad;
    float acc = 0.f;
    if (c < Creal) {
        acc = bias[c];
#pragma unroll
        for (int ky = 0; ky < 3; ++ky) {
            int hh = h + ky - 1;
            if (hh < 0 || hh > 127) continue;
#pragma unroll
            for (int kx = 0; kx < 3; ++kx) {
                int ww = x + kx - 1;
                if (ww < 0 || ww > 127) continue;
                acc += w[c * 9 + ky * 3 + kx] *
                       bf2f(in[base + (long)(hh * 128 + ww) * Cpad + c]);
            }
        }
    }
    out[base + (long)n * Cpad + c] = f2b(acc);
}

// ---------------------------------------------------------------------------
// Per-channel sum of squares over pixels for q (ch 0..255) / k (ch 256..511)
// ---------------------------------------------------------------------------
__global__ __launch_bounds__(256) void colsumsq_kernel(const unsigned short* __restrict__ qkv,
                                                       float* __restrict__ sumsq) {
    int b = blockIdx.z;
    int c = blockIdx.y * 256 + threadIdx.x;   // 0..511
    int n0 = blockIdx.x * 128;
    const unsigned short* p = qkv + ((long)b * HW_ + n0) * 768 + c;
    float s = 0.f;
    for (int i = 0; i < 128; ++i) {
        float v = bf2f(p[(long)i * 768]);
        s += v * v;
    }
    atomicAdd(&sumsq[b * 512 + c], s);
}

// ---------------------------------------------------------------------------
// Channel attention: S = temp * (q_n k_n^T) with l2 row scales, then softmax.
// One block per (head, batch).  P: (B*8, 32, 32) f32
// ---------------------------------------------------------------------------
__global__ __launch_bounds__(256) void attn_kernel(const unsigned short* __restrict__ qkvdw,
                                                   const float* __restrict__ sumsq,
                                                   const float* __restrict__ temp,
                                                   float* __restrict__ P) {
    int h = blockIdx.x;
    int b = blockIdx.y;
    __shared__ float qs[64][32];
    __shared__ float ks[64][32];
    __shared__ float S[32][32];
    int t = threadIdx.x;
    int c  = t >> 3;
    int d0 = (t & 7) * 4;
    float a0 = 0.f, a1 = 0.f, a2 = 0.f, a3 = 0.f;
    long base = (long)b * HW_;
    for (int n0 = 0; n0 < HW_; n0 += 64) {
        __syncthreads();
#pragma unroll
        for (int i = 0; i < 8; ++i) {
            int e = t + 256 * i;
            int nn = e >> 5, cc = e & 31;
            long rowoff = (base + n0 + nn) * 768;
            qs[nn][cc] = bf2f(qkvdw[rowoff + h * 32 + cc]);
            ks[nn][cc] = bf2f(qkvdw[rowoff + 256 + h * 32 + cc]);
        }
        __syncthreads();
#pragma unroll 4
        for (int nn = 0; nn < 64; ++nn) {
            float qv = qs[nn][c];
            a0 += qv * ks[nn][d0 + 0];
            a1 += qv * ks[nn][d0 + 1];
            a2 += qv * ks[nn][d0 + 2];
            a3 += qv * ks[nn][d0 + 3];
        }
    }
    float qsc = 1.f / fmaxf(sqrtf(sumsq[b * 512 + h * 32 + c]), 1e-12f);
    float tm  = temp[h] * qsc;
    const float* ksq = &sumsq[b * 512 + 256 + h * 32];
    S[c][d0 + 0] = a0 * tm / fmaxf(sqrtf(ksq[d0 + 0]), 1e-12f);
    S[c][d0 + 1] = a1 * tm / fmaxf(sqrtf(ksq[d0 + 1]), 1e-12f);
    S[c][d0 + 2] = a2 * tm / fmaxf(sqrtf(ksq[d0 + 2]), 1e-12f);
    S[c][d0 + 3] = a3 * tm / fmaxf(sqrtf(ksq[d0 + 3]), 1e-12f);
    __syncthreads();
    if (t < 32) {
        float m = -3.4e38f;
        for (int d = 0; d < 32; ++d) m = fmaxf(m, S[t][d]);
        float sum = 0.f;
        for (int d = 0; d < 32; ++d) { float e = expf(S[t][d] - m); S[t][d] = e; sum += e; }
        float r = 1.f / sum;
        float* Pp = P + ((long)(b * 8 + h) * 32 + t) * 32;
        for (int d = 0; d < 32; ++d) Pp[d] = S[t][d] * r;
    }
}

// ---------------------------------------------------------------------------
// out[n][c] = sum_d P[b,h][c%32][d] * v[n][512 + 32h + d]   (NHWC bf16 out)
// ---------------------------------------------------------------------------
__global__ __launch_bounds__(256) void attnv_kernel(const unsigned short* __restrict__ qkvdw,
                                                    const float* __restrict__ P,
                                                    unsigned short* __restrict__ out) {
    __shared__ float Ps[256 * 32];
    int b = blockIdx.y;
    int n = blockIdx.x;
    int t = threadIdx.x;
    int h = t >> 5, cc = t & 31;
    const float* Pp = P + ((long)(b * 8 + h) * 32 + cc) * 32;
#pragma unroll 8
    for (int d = 0; d < 32; ++d) Ps[t * 32 + d] = Pp[d];
    __syncthreads();
    long row = (long)b * HW_ + n;
    const unsigned short* v = qkvdw + row * 768 + 512 + h * 32;
    float acc = 0.f;
#pragma unroll 8
    for (int d = 0; d < 32; ++d) acc += Ps[t * 32 + d] * bf2f(v[d]);
    out[row * CDIM + t] = f2b(acc);
}

// ---------------------------------------------------------------------------
// Gated GELU: g[r][c] = y1 * gelu(y2), c<680; pad channels (680..703) = 0
// ---------------------------------------------------------------------------
__global__ __launch_bounds__(256) void gate_kernel(const unsigned short* __restrict__ dw2,
                                                   unsigned short* __restrict__ g) {
    long i = (long)blockIdx.x * 256 + threadIdx.x;
    if (i >= (long)65536 * 704) return;
    long r = i / 704;
    int  c = (int)(i % 704);
    float val = 0.f;
    if (c < 680) {
        float y1 = bf2f(dw2[r * 1408 + c]);
        float y2 = bf2f(dw2[r * 1408 + 680 + c]);
        val = y1 * (0.5f * y2 * (1.f + erff(y2 * 0.70710678118654752f)));
    }
    g[i] = f2b(val);
}

// ---------------------------------------------------------------------------
// Host launcher
// ---------------------------------------------------------------------------
extern "C" void kernel_launch(void* const* d_in, const int* in_sizes, int n_in,
                              void* d_out, int out_size, void* d_ws, size_t ws_size,
                              hipStream_t stream) {
    (void)in_sizes; (void)n_in; (void)out_size; (void)ws_size;
    const int Bn = 4, C = 256, HW = 16384, R = 65536;
    const int C3 = 768, H2 = 1360, H2P = 1408, HID = 680, HP = 704;

    const float* x       = (const float*)d_in[0];
    const float* n1_w    = (const float*)d_in[1];
    const float* n1_b    = (const float*)d_in[2];
    const float* temp    = (const float*)d_in[3];
    const float* qkv_w   = (const float*)d_in[4];
    const float* qkv_b   = (const float*)d_in[5];
    const float* qkv_dww = (const float*)d_in[6];
    const float* qkv_dwb = (const float*)d_in[7];
    const float* po_w    = (const float*)d_in[8];
    const float* po_b    = (const float*)d_in[9];
    const float* n2_w    = (const float*)d_in[10];
    const float* n2_b    = (const float*)d_in[11];
    const float* pin_w   = (const float*)d_in[12];
    const float* pin_b   = (const float*)d_in[13];
    const float* dw_w    = (const float*)d_in[14];
    const float* dw_b    = (const float*)d_in[15];
    const float* pout_w  = (const float*)d_in[16];
    const float* pout_b  = (const float*)d_in[17];

    char* ws = (char*)d_ws;
    size_t off = 0;
    auto alloc = [&](size_t bytes) { size_t o = off; off += (bytes + 255) & ~(size_t)255; return o; };

    unsigned short* ln_bf   = (unsigned short*)(ws + alloc((size_t)R * C * 2));       // 32 MB
    size_t reg2             = alloc((size_t)R * C3 * 2 * 2);                          // 192 MB shared
    unsigned short* qkv_g   = (unsigned short*)(ws + reg2);
    unsigned short* qkv_dw  = (unsigned short*)(ws + reg2 + (size_t)R * C3 * 2);
    unsigned short* pin_out = (unsigned short*)(ws + reg2);                           // reuse (176 MB)
    unsigned short* gbuf    = (unsigned short*)(ws + reg2);                           // reuse (88 MB)
    unsigned short* dw2     = (unsigned short*)(ws + alloc((size_t)R * H2P * 2));     // 176 MB
    unsigned short* attnout = (unsigned short*)(ws + alloc((size_t)R * C * 2));       // 32 MB
    float*          x1      = (float*)(ws + alloc((size_t)R * C * 4));                // 64 MB
    unsigned short* wq      = (unsigned short*)(ws + alloc((size_t)C3 * C * 2));
    unsigned short* wpo     = (unsigned short*)(ws + alloc((size_t)C * C * 2));
    unsigned short* wpin    = (unsigned short*)(ws + alloc((size_t)H2P * C * 2));
    unsigned short* wpout   = (unsigned short*)(ws + alloc((size_t)C * HP * 2));
    float*          bpin    = (float*)(ws + alloc((size_t)H2P * 4));
    float*          sumsq   = (float*)(ws + alloc((size_t)Bn * 512 * 4));
    float*          P       = (float*)(ws + alloc((size_t)Bn * 8 * 32 * 32 * 4));

    dim3 blk(256);

    // weight/bias prep (bf16, zero-padded)
    convert_weight_kernel<<<dim3((C3 * C + 255) / 256), blk, 0, stream>>>(qkv_w, wq, C3, C, C3, C);
    convert_weight_kernel<<<dim3((C * C + 255) / 256), blk, 0, stream>>>(po_w, wpo, C, C, C, C);
    convert_weight_kernel<<<dim3((H2P * C + 255) / 256), blk, 0, stream>>>(pin_w, wpin, H2, C, H2P, C);
    convert_weight_kernel<<<dim3((C * HP + 255) / 256), blk, 0, stream>>>(pout_w, wpout, C, HID, C, HP);
    pad_bias_kernel<<<dim3((H2P + 255) / 256), blk, 0, stream>>>(pin_b, bpin, H2, H2P);
    zero_kernel<<<dim3((Bn * 512 + 255) / 256), blk, 0, stream>>>(sumsq, Bn * 512);

    // --- attention branch ---
    ln_kernel<<<dim3(HW / 256, Bn), blk, 0, stream>>>(x, n1_w, n1_b, ln_bf);
    gemm_bf16<0><<<dim3(R / 64, C3 / 128), blk, 0, stream>>>(ln_bf, wq, qkv_b, qkv_g,
                                                             nullptr, nullptr, C, C3, C3);
    dwconv_kernel<<<dim3(HW, C3 / 256, Bn), blk, 0, stream>>>(qkv_g, qkv_dww, qkv_dwb,
                                                              qkv_dw, C3, C3);
    colsumsq_kernel<<<dim3(HW / 128, 2, Bn), blk, 0, stream>>>(qkv_dw, sumsq);
    attn_kernel<<<dim3(8, Bn), blk, 0, stream>>>(qkv_dw, sumsq, temp, P);
    attnv_kernel<<<dim3(HW, Bn), blk, 0, stream>>>(qkv_dw, P, attnout);
    gemm_bf16<1><<<dim3(R / 64, C / 128), blk, 0, stream>>>(attnout, wpo, po_b,
                                                            nullptr, x1, x, C, C, C);

    // --- FFN branch ---
    ln_kernel<<<dim3(HW / 256, Bn), blk, 0, stream>>>(x1, n2_w, n2_b, ln_bf);
    gemm_bf16<0><<<dim3(R / 64, H2P / 128), blk, 0, stream>>>(ln_bf, wpin, bpin, pin_out,
                                                              nullptr, nullptr, C, H2P, H2P);
    dwconv_kernel<<<dim3(HW, (H2P + 255) / 256, Bn), blk, 0, stream>>>(pin_out, dw_w, dw_b,
                                                                       dw2, H2P, H2);
    gate_kernel<<<dim3((int)(((size_t)R * HP + 255) / 256)), blk, 0, stream>>>(dw2, gbuf);
    gemm_bf16<1><<<dim3(R / 64, C / 128), blk, 0, stream>>>(gbuf, wpout, pout_b,
                                                            nullptr, (float*)d_out, x1, HP, C, C);
}
